// MyRNN_27049704030526
// MI455X (gfx1250) — matrix operand order repaired and checked by reading, not compiled, not run on hardware
//
#include <hip/hip_runtime.h>

#define S_LEN 512
#define BATCH 64
#define ISZ   512
#define HSZ   1024

typedef __attribute__((ext_vector_type(16))) __bf16   v16bf;
typedef __attribute__((ext_vector_type(8)))  float    v8f;
typedef __attribute__((ext_vector_type(4)))  unsigned u32x4;
typedef __attribute__((ext_vector_type(4)))  int      i32x4;
typedef __attribute__((ext_vector_type(8)))  int      i32x8;
typedef __attribute__((ext_vector_type(4)))  float    f32x4;

union Frag { u32x4 u[2]; v16bf v; };

// ---------------- CDNA5 async global->LDS + TDM paths (guarded) ------------
#if defined(__has_builtin)
#if __has_builtin(__builtin_amdgcn_global_load_async_to_lds_b128)
#define HAVE_ASYNC_LDS 1
#endif
#if __has_builtin(__builtin_amdgcn_tensor_load_to_lds)
#define HAVE_TDM 1
#endif
#endif

__device__ __forceinline__ void g2l_copy16(void* lds_dst, const void* gsrc) {
#if HAVE_ASYNC_LDS
  typedef __attribute__((address_space(1))) i32x4* gp_t;
  typedef __attribute__((address_space(3))) i32x4* lp_t;
  __builtin_amdgcn_global_load_async_to_lds_b128(
      (gp_t)const_cast<void*>(gsrc), (lp_t)lds_dst, 0, 0);
#else
  *(u32x4*)lds_dst = *(const u32x4*)gsrc;
#endif
}

__device__ __forceinline__ void g2l_wait() {
#if HAVE_ASYNC_LDS
#if __has_builtin(__builtin_amdgcn_s_wait_asynccnt)
  __builtin_amdgcn_s_wait_asynccnt(0);
#else
  asm volatile("s_wait_asynccnt 0" ::: "memory");
#endif
#endif
}

__device__ __forceinline__ void tdm_wait() {
#if __has_builtin(__builtin_amdgcn_s_wait_tensorcnt)
  __builtin_amdgcn_s_wait_tensorcnt(0);
#else
  asm volatile("s_wait_tensorcnt 0" ::: "memory");
#endif
}

// Padded LDS layout used by TDM pad engine: +8 elems after every 512 elems
// (pad_interval=7 -> 256 dwords, pad_amount=3 -> 4 dwords). Row stride = 1040.
#define WLD 1040
#define PADK(k) ((k) + ((((unsigned)(k)) >> 9) << 3))

// One-shot DMA of a [64 x 1024] bf16 tile (row stride 1024 elems in memory)
// into LDS with the padded layout above. TDM descriptor per ISA 08 section 8.
__device__ __forceinline__ void load_64x1024_to_lds(__bf16* lds_arr,
                                                    const __bf16* g, int t) {
#if HAVE_TDM
  if (t < 32) {  // one TDM op per workgroup (wave 0); EXEC ignored by TDM
    unsigned lds_off = (unsigned)(uintptr_t)
        (__attribute__((address_space(3))) __bf16*)lds_arr;
    unsigned long long ga = (unsigned long long)(uintptr_t)g;
    // Group 0: count=1 | lds_addr | global_addr[56:0] | type=2 ("image")
    u32x4 g0 = { 1u,
                 lds_off,
                 (unsigned)ga,
                 (unsigned)((ga >> 32) & 0x01FFFFFFu) | (2u << 30) };
    // Group 1: wg_mask=0, data_size=1(2B), pad_enable, pad_interval=7(256dw),
    // pad_amount=3(4dw); tensor_dim0=1024, tensor_dim1=64;
    // tile_dim0=1024, tile_dim1=64; tensor_dim0_stride=1024.
    i32x8 g1 = { (int)((1u << 16) | (1u << 20) | (7u << 22) | (3u << 25)),
                 (int)((1024u & 0xFFFFu) << 16),  // tensor_dim0[15:0] @ [31:16]
                 (int)((64u & 0xFFFFu) << 16),    // dim0 hi=0 | tensor_dim1 lo
                 (int)(1024u << 16),              // dim1 hi=0 | tile_dim0
                 (int)64,                         // tile_dim1=64, tile_dim2=0
                 (int)1024,                       // tensor_dim0_stride lo32
                 0, 0 };                          // stride hi, dim1_stride=0
    i32x4 gz4 = { 0, 0, 0, 0 };                   // groups 2/3 unused (2D)
    i32x8 gz8 = { 0, 0, 0, 0, 0, 0, 0, 0 };
    __builtin_amdgcn_tensor_load_to_lds(g0, g1, gz4, gz4, gz8, 0);
    tdm_wait();
  }
#else
  for (int idx = t; idx < 64 * 128; idx += 256) {
    const int row = idx >> 7, c = (idx & 127) * 8;
    g2l_copy16(&lds_arr[row * WLD + PADK(c)], g + (size_t)row * 1024 + c);
  }
  g2l_wait();
#endif
}

__device__ __forceinline__ v8f wmma_bf16(const Frag& a, const Frag& b, v8f c) {
  return __builtin_amdgcn_wmma_f32_16x16x32_bf16(
      false, a.v, false, b.v, (short)0, c, false, false);
}

// ---------------- Kernel 1: fp32 -> bf16 weight conversion + barrier reset --
__global__ void cvt_kernel(const float* __restrict__ Wih,
                           const float* __restrict__ Whh,
                           const float* __restrict__ bih,
                           const float* __restrict__ bhh,
                           __bf16* __restrict__ Wih_b,
                           __bf16* __restrict__ Whh_b,
                           float*  __restrict__ bsum,
                           unsigned* __restrict__ bar) {
  size_t tid = (size_t)blockIdx.x * blockDim.x + threadIdx.x;
  size_t nt  = (size_t)gridDim.x * blockDim.x;
  if (tid == 0) *bar = 0u;  // deterministic across graph replays
  for (size_t i = tid; i < (size_t)HSZ * ISZ; i += nt) Wih_b[i] = (__bf16)Wih[i];
  for (size_t i = tid; i < (size_t)HSZ * HSZ; i += nt) Whh_b[i] = (__bf16)Whh[i];
  for (size_t i = tid; i < HSZ; i += nt) bsum[i] = bih[i] + bhh[i];
}

// ---------------- Kernel 2: x_proj = x @ Wih^T + bias  ([32768,512]x[512,1024])
#define BM 128
#define BN 128
#define BK 32
#define LDT 40  // padded LDS row stride (elements) to dodge bank conflicts

__global__ __launch_bounds__(256) void xproj_kernel(
    const float*  __restrict__ x,      // [S*B, I]
    const __bf16* __restrict__ Wih_b,  // [H, I]
    const float*  __restrict__ bsum,   // [H]
    float*        __restrict__ out)    // [S*B, H]
{
  __shared__ __bf16 As[BM * LDT];
  __shared__ __bf16 Bs[BN * LDT];

  const int tileM = blockIdx.x * BM;
  const int tileN = blockIdx.y * BN;
  const int t = threadIdx.x;
  const int lane = t & 31, wid = t >> 5;
  const int wm = wid >> 2;  // 0..1 -> 64 rows each
  const int wn = wid & 3;   // 0..3 -> 32 cols each
  const int r = lane & 15, hs = lane >> 4;
  const int lrow = t >> 1, lseg = t & 1;  // cooperative tile-load mapping

  v8f acc[4][2];
  const v8f vzero = {0.f, 0.f, 0.f, 0.f, 0.f, 0.f, 0.f, 0.f};
  for (int i = 0; i < 4; ++i)
    for (int j = 0; j < 2; ++j) acc[i][j] = vzero;

  for (int k0 = 0; k0 < ISZ; k0 += BK) {
    // A tile: 128x32 fp32 -> bf16 into LDS (2 threads/row, 16 floats each)
    {
      const f32x4* src = (const f32x4*)(x + (size_t)(tileM + lrow) * ISZ + k0 + lseg * 16);
      f32x4 f0 = src[0], f1 = src[1], f2 = src[2], f3 = src[3];
      __bf16* dst = &As[lrow * LDT + lseg * 16];
      dst[0]  = (__bf16)f0.x; dst[1]  = (__bf16)f0.y; dst[2]  = (__bf16)f0.z; dst[3]  = (__bf16)f0.w;
      dst[4]  = (__bf16)f1.x; dst[5]  = (__bf16)f1.y; dst[6]  = (__bf16)f1.z; dst[7]  = (__bf16)f1.w;
      dst[8]  = (__bf16)f2.x; dst[9]  = (__bf16)f2.y; dst[10] = (__bf16)f2.z; dst[11] = (__bf16)f2.w;
      dst[12] = (__bf16)f3.x; dst[13] = (__bf16)f3.y; dst[14] = (__bf16)f3.z; dst[15] = (__bf16)f3.w;
    }
    // B tile: 128x32 bf16, async -> LDS (2x16B per thread)
    {
      const __bf16* src = Wih_b + (size_t)(tileN + lrow) * ISZ + k0 + lseg * 16;
      g2l_copy16(&Bs[lrow * LDT + lseg * 16],     src);
      g2l_copy16(&Bs[lrow * LDT + lseg * 16 + 8], src + 8);
    }
    g2l_wait();
    __syncthreads();

    Frag a[4], b[2];
    for (int i = 0; i < 4; ++i) {  // A frag: rows wm*64+i*16.., contiguous k = 16*hs + s per lane-half
      const __bf16* p = &As[(wm * 64 + i * 16 + r) * LDT + hs * 16];
      a[i].u[0] = *(const u32x4*)p;
      a[i].u[1] = *(const u32x4*)(p + 8);
    }
    for (int j = 0; j < 2; ++j) {  // B frag: cols wn*32+j*16.., K group by lane-half
      const __bf16* p = &Bs[(wn * 32 + j * 16 + r) * LDT + hs * 16];
      b[j].u[0] = *(const u32x4*)p;
      b[j].u[1] = *(const u32x4*)(p + 8);
    }
    for (int i = 0; i < 4; ++i)
      for (int j = 0; j < 2; ++j) acc[i][j] = wmma_bf16(a[i], b[j], acc[i][j]);
    __syncthreads();
  }

  for (int j = 0; j < 2; ++j) {
    const int n = tileN + wn * 32 + j * 16 + r;
    const float bias = bsum[n];
    for (int i = 0; i < 4; ++i)
      for (int v = 0; v < 8; ++v) {
        const int m = tileM + wm * 64 + i * 16 + v + 8 * hs;
        out[(size_t)m * HSZ + n] = acc[i][j][v] + bias;
      }
  }
}

// ---------------- Kernel 3: persistent recurrent scan ----------------------
// 16 workgroups, one per WGP; block j owns H columns [j*64, j*64+64).
// W_hh slice (64x1024 bf16 ~ 130KB padded) + full h (~130KB padded) in LDS;
// both tiles filled by the Tensor Data Mover.
__global__ __launch_bounds__(256) void rnn_scan_kernel(
    const __bf16* __restrict__ Whh_b,  // [H, H]
    float*        __restrict__ out,    // [S, B, H]: holds x_proj, overwritten with h_s
    float*        __restrict__ hlast,  // [B, H]
    __bf16*       __restrict__ hg,     // [2][B, H] double-buffered staging
    unsigned*     __restrict__ bar)
{
  __shared__ __bf16 Whh_s[64 * WLD];
  __shared__ __bf16 h_s[BATCH * WLD];

  const int j = blockIdx.x;
  const int t = threadIdx.x;
  const int lane = t & 31, wid = t >> 5;
  const int wm = wid & 3;        // batch rows [wm*16, +16)
  const int wn = wid >> 2;       // 0..1: cols [wn*32, +32)
  const int r = lane & 15, hs = lane >> 4;

  // TDM-load W_hh slice; init h_s = h0 = 0 (zero pads too).
  load_64x1024_to_lds(Whh_s, Whh_b + (size_t)j * 64 * HSZ, t);
  const u32x4 z4 = {0u, 0u, 0u, 0u};
  for (int idx = t; idx < BATCH * (WLD / 8); idx += 256)
    *(u32x4*)&h_s[idx * 8] = z4;
  __syncthreads();

  const v8f vzero = {0.f, 0.f, 0.f, 0.f, 0.f, 0.f, 0.f, 0.f};

  for (int s = 0; s < S_LEN; ++s) {
    v8f acc0 = vzero, acc1 = vzero;
    for (int k0 = 0; k0 < HSZ; k0 += 32) {
      Frag a, b0, b1;
      const int kA = k0 + hs * 16;
      a.u[0] = *(const u32x4*)&h_s[(wm * 16 + r) * WLD + PADK(kA)];
      a.u[1] = *(const u32x4*)&h_s[(wm * 16 + r) * WLD + PADK(kA + 8)];
      const int kB = k0 + hs * 16;  // 16-elem chunk never crosses a 512 boundary
      const __bf16* pb0 = &Whh_s[(wn * 32 + r) * WLD + PADK(kB)];
      b0.u[0] = *(const u32x4*)pb0;
      b0.u[1] = *(const u32x4*)(pb0 + 8);
      const __bf16* pb1 = &Whh_s[(wn * 32 + 16 + r) * WLD + PADK(kB)];
      b1.u[0] = *(const u32x4*)pb1;
      b1.u[1] = *(const u32x4*)(pb1 + 8);
      acc0 = wmma_bf16(a, b0, acc0);
      acc1 = wmma_bf16(a, b1, acc1);
    }

    // Epilogue: h = tanh(x_proj + h@Whh^T); write out (in place) + bf16 stage
    __bf16* hg_cur = hg + (size_t)(s & 1) * BATCH * HSZ;
    for (int nsub = 0; nsub < 2; ++nsub) {
      const int n = j * 64 + wn * 32 + nsub * 16 + r;
      const v8f acc = nsub ? acc1 : acc0;
      for (int v = 0; v < 8; ++v) {
        const int brow = wm * 16 + v + 8 * hs;
        const size_t oidx = ((size_t)s * BATCH + brow) * HSZ + n;
        const float val = tanhf(out[oidx] + acc[v]);
        out[oidx] = val;
        hg_cur[(size_t)brow * HSZ + n] = (__bf16)val;
        if (s == S_LEN - 1) hlast[(size_t)brow * HSZ + n] = val;
      }
    }

    // Grid barrier (device-scope atomic, monotonic generation counter)
    __threadfence();
    __syncthreads();
    if (t == 0) {
      __hip_atomic_fetch_add(bar, 1u, __ATOMIC_ACQ_REL, __HIP_MEMORY_SCOPE_AGENT);
      const unsigned target = (unsigned)(s + 1) * (unsigned)gridDim.x;
      while (__hip_atomic_load(bar, __ATOMIC_ACQUIRE, __HIP_MEMORY_SCOPE_AGENT) < target)
        __builtin_amdgcn_s_sleep(1);
    }
    __syncthreads();

    // Pull the full new h back into LDS with one TDM op (wave 0)
    load_64x1024_to_lds(h_s, hg_cur, t);
    __syncthreads();
  }
}

// ---------------- Host launcher --------------------------------------------
extern "C" void kernel_launch(void* const* d_in, const int* in_sizes, int n_in,
                              void* d_out, int out_size, void* d_ws, size_t ws_size,
                              hipStream_t stream) {
  const float* x   = (const float*)d_in[0];
  const float* Wih = (const float*)d_in[1];
  const float* Whh = (const float*)d_in[2];
  const float* bih = (const float*)d_in[3];
  const float* bhh = (const float*)d_in[4];

  float* out   = (float*)d_out;                       // [S,B,H]
  float* hlast = out + (size_t)S_LEN * BATCH * HSZ;   // [B,H]

  char* ws = (char*)d_ws;
  unsigned* bar  = (unsigned*)ws;                             // 256 B reserved
  __bf16* Wih_b  = (__bf16*)(ws + 256);                       // 1 MB
  __bf16* Whh_b  = Wih_b + (size_t)HSZ * ISZ;                 // 2 MB
  float*  bsum   = (float*)(Whh_b + (size_t)HSZ * HSZ);       // 4 KB
  __bf16* hg     = (__bf16*)(bsum + HSZ);                     // 2 x 128 KB

  cvt_kernel<<<512, 256, 0, stream>>>(Wih, Whh, bih, bhh, Wih_b, Whh_b, bsum, bar);

  dim3 g1((S_LEN * BATCH) / BM, HSZ / BN);  // (256, 8)
  xproj_kernel<<<g1, 256, 0, stream>>>(x, Wih_b, bsum, out);

  rnn_scan_kernel<<<16, 256, 0, stream>>>(Whh_b, out, hlast, hg, bar);
}
